// LlamaAttention_59124519796894
// MI455X (gfx1250) — 8-bit matrix operand order repaired and checked by reading, compiled, not run on hardware
//
#include <hip/hip_runtime.h>
#include <hip/hip_bf16.h>
#include <math.h>

constexpr int Bsz   = 32;
constexpr int Hh    = 32;
constexpr int HKVh  = 8;
constexpr int Dd    = 128;
constexpr int HIDn  = 4096;
constexpr int KVn   = 4096;
constexpr int Gg    = Hh / HKVh;
constexpr int CHUNKn= 32;
constexpr int NCn   = KVn / CHUNKn;
constexpr int BUDGET= 1024 / CHUNKn;
constexpr int SINKC = 1;
constexpr int LOCALC= 2;
constexpr int QKVW  = (Hh + 2 * HKVh) * Dd;
#define SCALING 0.08838834764831845f

typedef __attribute__((ext_vector_type(8))) int v8i;

__global__ __launch_bounds__(256) void row_quant_kernel(
    const float* __restrict__ in, unsigned* __restrict__ qout,
    float* __restrict__ sout, int width)
{
  int row = blockIdx.x;
  const float* r = in + (size_t)row * width;
  float amax = 0.f;
  for (int i = threadIdx.x; i < width; i += blockDim.x)
    amax = fmaxf(amax, fabsf(r[i]));
  #pragma unroll
  for (int off = 16; off; off >>= 1)
    amax = fmaxf(amax, __shfl_xor(amax, off, 32));
  __shared__ float red[8];
  int wave = threadIdx.x >> 5, lane = threadIdx.x & 31;
  if (lane == 0) red[wave] = amax;
  __syncthreads();
  if (threadIdx.x == 0) {
    float m = red[0];
    for (int i = 1; i < 8; ++i) m = fmaxf(m, red[i]);
    red[0] = fmaxf(m * (1.0f / 127.0f), 1e-8f);
  }
  __syncthreads();
  float s = red[0];
  if (threadIdx.x == 0) sout[row] = s;
  float inv = 1.0f / s;
  for (int i = threadIdx.x * 4; i < width; i += blockDim.x * 4) {
    unsigned w = 0;
    #pragma unroll
    for (int j = 0; j < 4; ++j) {
      float v = rintf(r[i + j] * inv);
      v = fminf(fmaxf(v, -127.f), 127.f);
      w |= ((unsigned)(((int)v) & 0xFF)) << (8 * j);
    }
    qout[((size_t)row * width + i) >> 2] = w;
  }
}

__device__ __forceinline__ unsigned pack4(const int4& ww) {
  unsigned t0 = __builtin_amdgcn_perm((unsigned)ww.y, (unsigned)ww.x, 0x0C0C0400u);
  unsigned t1 = __builtin_amdgcn_perm((unsigned)ww.w, (unsigned)ww.z, 0x04000C0Cu);
  return t0 | t1;
}

__device__ __forceinline__ v8i load_a_block(const unsigned* __restrict__ arow,
                                            int k0, int half) {
  v8i a;
  #pragma unroll
  for (int vp = 0; vp < 4; ++vp) {
    int koff = k0 + ((vp & 1) << 3) + ((vp >> 1) << 5) + (half << 4);
    uint2 av = *(const uint2*)(arow + (koff >> 2));
    a[2 * vp]     = (int)av.x;
    a[2 * vp + 1] = (int)av.y;
  }
  return a;
}

__device__ __forceinline__ v8i load_b_block(const int4* __restrict__ wrow4,
                                            int k0, int half) {
  v8i b;
  #pragma unroll
  for (int v = 0; v < 8; ++v) {
    int kb = k0 + ((v >> 2) << 5) + (half << 4) + ((v & 3) << 2);
    b[v] = (int)pack4(wrow4[kb >> 2]);
  }
  return b;
}

__global__ __launch_bounds__(32) void gemm_w8a8_kernel(
    const unsigned* __restrict__ aq,
    const int*      __restrict__ w,
    const float*    __restrict__ ascale,
    const float*    __restrict__ wscale,
    float*          __restrict__ out,
    int M, int N, int K)
{
  int ntiles = N >> 4;
  int tm = blockIdx.x / ntiles;
  int tn = blockIdx.x % ntiles;
  int lane = threadIdx.x;
  int half = lane >> 4;
  int l15  = lane & 15;

  v8i acc = {};
  int m = tm * 16 + l15;
  int n = tn * 16 + l15;
  const unsigned* arow = aq + (size_t)m * (K >> 2);
  const int4* wrow4 = (const int4*)(w + (size_t)n * K);

  for (int k0 = 0; k0 < K; k0 += 128) {
    if (k0 + 128 < K)
      __builtin_prefetch((const void*)(wrow4 + ((k0 + 128) >> 2)), 0, 3);
    v8i a0 = load_a_block(arow, k0, half);
    v8i b0 = load_b_block(wrow4, k0, half);
    v8i a1 = load_a_block(arow, k0 + 64, half);
    v8i b1 = load_b_block(wrow4, k0 + 64, half);
    acc = __builtin_amdgcn_wmma_i32_16x16x64_iu8(true, a0, true, b0, acc, false, false);
    acc = __builtin_amdgcn_wmma_i32_16x16x64_iu8(true, a1, true, b1, acc, false, false);
  }
  #pragma unroll
  for (int r = 0; r < 8; ++r) {
    int mr = tm * 16 + r + half * 8;
    int nc = tn * 16 + l15;
    out[(size_t)mr * N + nc] = (float)acc[r] * ascale[mr] * wscale[nc];
  }
}

__global__ __launch_bounds__(64) void rope_kv_kernel(
    float* __restrict__ qkv, float* __restrict__ kd, float* __restrict__ vd,
    const float* __restrict__ kv_scale)
{
  int b = blockIdx.x / (Hh + 2 * HKVh);
  int h = blockIdx.x % (Hh + 2 * HKVh);
  float* row = qkv + (size_t)b * QKVW + h * Dd;
  int i = threadIdx.x;
  if (h < Hh + HKVh) {
    float inv = powf(10000.f, -(float)i / 64.f);
    float ang = 4096.f * inv;
    float c = cosf(ang), s = sinf(ang);
    float x1 = row[i], x2 = row[i + 64];
    float r1 = x1 * c - x2 * s;
    float r2 = x2 * c + x1 * s;
    row[i] = r1; row[i + 64] = r2;
    if (h >= Hh) {
      float s0 = kv_scale[0];
      float* kr = kd + ((size_t)b * HKVh + (h - Hh)) * Dd;
      kr[i]      = fminf(fmaxf(rintf(r1 / s0), -127.f), 127.f) * s0;
      kr[i + 64] = fminf(fmaxf(rintf(r2 / s0), -127.f), 127.f) * s0;
    }
  } else {
    float s1 = kv_scale[1];
    float* vr = vd + ((size_t)b * HKVh + (h - Hh - HKVh)) * Dd;
    float x1 = row[i], x2 = row[i + 64];
    vr[i]      = fminf(fmaxf(rintf(x1 / s1), -127.f), 127.f) * s1;
    vr[i + 64] = fminf(fmaxf(rintf(x2 / s1), -127.f), 127.f) * s1;
  }
}

__global__ __launch_bounds__(128) void chunk_score_kernel(
    const int* __restrict__ kcache, const float* __restrict__ qkv,
    const float* __restrict__ kv_scale, float* __restrict__ score)
{
  int blk = blockIdx.x;
  int c  = blk % NCn;
  int kv = (blk / NCn) % HKVh;
  int b  = blk / (NCn * HKVh);
  int d  = threadIdx.x;
  const int* base = kcache + (((size_t)b * KVn + (size_t)c * CHUNKn) * HKVh + kv) * Dd + d;
  int mn = base[0], mx = base[0];
  for (int t = 1; t < CHUNKn; ++t) {
    int v = base[(size_t)t * HKVh * Dd];
    mn = min(mn, v); mx = max(mx, v);
  }
  float s0 = kv_scale[0];
  float fmx = mx * s0, fmn = mn * s0;
  __shared__ float red[4];
  int wave = d >> 5, lane = d & 31;
  for (int g = 0; g < Gg; ++g) {
    float q = qkv[(size_t)b * QKVW + (kv * Gg + g) * Dd + d];
    float p = fmaxf(q * fmx, q * fmn);
    #pragma unroll
    for (int off = 16; off; off >>= 1) p += __shfl_xor(p, off, 32);
    if (lane == 0) red[wave] = p;
    __syncthreads();
    if (d == 0)
      score[(((size_t)b * HKVh + kv) * Gg + g) * NCn + c] = red[0] + red[1] + red[2] + red[3];
    __syncthreads();
  }
}

__global__ __launch_bounds__(128) void topk_kernel(
    const float* __restrict__ score, unsigned* __restrict__ sel)
{
  int row = blockIdx.x;
  int c = threadIdx.x;
  __shared__ float sc[NCn];
  __shared__ int taken[NCn];
  float v = score[(size_t)row * NCn + c];
  if (c < SINKC || c >= NCn - LOCALC) v = 3e38f;
  sc[c] = v; taken[c] = 0;
  __syncthreads();
  if (c == 0) {
    for (int it = 0; it < BUDGET; ++it) {
      float best = -3.4e38f; int bi = 0;
      for (int i = 0; i < NCn; ++i)
        if (!taken[i] && sc[i] > best) { best = sc[i]; bi = i; }
      taken[bi] = 1;
    }
  }
  __syncthreads();
  if (c < 4) {
    unsigned m = 0;
    for (int i = 0; i < 32; ++i) if (taken[c * 32 + i]) m |= 1u << i;
    sel[(size_t)row * 4 + c] = m;
  }
}

__global__ __launch_bounds__(128) void attn_kernel(
    const int* __restrict__ kcache, const int* __restrict__ vcache,
    const float* __restrict__ qkv, const float* __restrict__ kd,
    const float* __restrict__ vd, const unsigned* __restrict__ sel,
    const float* __restrict__ kv_scale, float* __restrict__ attn_out)
{
  int row = blockIdx.x;
  int g  = row & (Gg - 1);
  int kv = (row >> 2) & (HKVh - 1);
  int b  = row >> 5;
  int d  = threadIdx.x;
  int lane = d & 31, wave = d >> 5;

  __shared__ float qs[Dd];
  __shared__ float lds_s[CHUNKn];

  qs[d] = qkv[(size_t)b * QKVW + (kv * Gg + g) * Dd + d] * SCALING;
  __syncthreads();

  float s0 = kv_scale[0], s1 = kv_scale[1];
  float m = -3.4e38f, l = 0.f, o = 0.f;
  const unsigned* srow = sel + (size_t)row * 4;
  unsigned selm[4] = { srow[0], srow[1], srow[2], srow[3] };

  const size_t tokstride = (size_t)HKVh * Dd;
  const int* kbase = kcache + ((size_t)b * KVn * HKVh + (size_t)kv) * Dd;
  const int* vbase = vcache + ((size_t)b * KVn * HKVh + (size_t)kv) * Dd;

  for (int c = 0; c < NCn; ++c) {
    if (!((selm[c >> 5] >> (c & 31)) & 1u)) continue;
    for (int tt = 0; tt < 8; ++tt) {
      int tok = c * CHUNKn + wave * 8 + tt;
      const int* kp = kbase + (size_t)tok * tokstride;
      float p = 0.f;
      #pragma unroll
      for (int dd = 0; dd < 4; ++dd) {
        int di = lane + dd * 32;
        p += qs[di] * (float)kp[di];
      }
      #pragma unroll
      for (int off = 16; off; off >>= 1) p += __shfl_xor(p, off, 32);
      if (lane == 0) lds_s[wave * 8 + tt] = p * s0;
    }
    __syncthreads();
    float cmax = lds_s[0];
    for (int t = 1; t < CHUNKn; ++t) cmax = fmaxf(cmax, lds_s[t]);
    float mn = fmaxf(m, cmax);
    float co = expf(m - mn);
    o *= co; l *= co;
    const int* vp = vbase + (size_t)(c * CHUNKn) * tokstride + d;
    for (int t = 0; t < CHUNKn; ++t) {
      float wgt = expf(lds_s[t] - mn);
      l += wgt;
      o += wgt * (float)vp[(size_t)t * tokstride];
    }
    m = mn;
    __syncthreads();
  }

  const float* kdr = kd + ((size_t)b * HKVh + kv) * Dd;
  float p = qs[d] * kdr[d];
  #pragma unroll
  for (int off = 16; off; off >>= 1) p += __shfl_xor(p, off, 32);
  if (lane == 0) lds_s[wave] = p;
  __syncthreads();
  float snew = lds_s[0] + lds_s[1] + lds_s[2] + lds_s[3];
  float mn = fmaxf(m, snew);
  float co = expf(m - mn);
  float wnew = expf(snew - mn);
  l = l * co + wnew;
  const float* vdr = vd + ((size_t)b * HKVh + kv) * Dd;
  float oval = o * co * s1 + wnew * vdr[d];
  attn_out[(size_t)b * HIDn + (kv * Gg + g) * Dd + d] = oval / l;
}

extern "C" void kernel_launch(void* const* d_in, const int* in_sizes, int n_in,
                              void* d_out, int out_size, void* d_ws, size_t ws_size,
                              hipStream_t stream) {
  (void)in_sizes; (void)n_in; (void)out_size; (void)ws_size;
  const float* x      = (const float*)d_in[0];
  const float* wqs    = (const float*)d_in[1];
  const float* wos    = (const float*)d_in[2];
  const float* kvs    = (const float*)d_in[3];
  const int*   wqkv   = (const int*)d_in[4];
  const int*   wo     = (const int*)d_in[5];
  const int*   kcache = (const int*)d_in[6];
  const int*   vcache = (const int*)d_in[7];
  float* y = (float*)d_out;

  char* p = (char*)d_ws;
  auto carve = [&](size_t bytes) -> char* {
    char* r = p; p += (bytes + 255) & ~(size_t)255; return r;
  };
  unsigned* xq    = (unsigned*)carve((size_t)Bsz * HIDn);
  float*    xs    = (float*)carve(Bsz * sizeof(float));
  float*    qkv   = (float*)carve((size_t)Bsz * QKVW * 4);
  float*    kd    = (float*)carve((size_t)Bsz * HKVh * Dd * 4);
  float*    vd    = (float*)carve((size_t)Bsz * HKVh * Dd * 4);
  float*    score = (float*)carve((size_t)Bsz * HKVh * Gg * NCn * 4);
  unsigned* sel   = (unsigned*)carve((size_t)Bsz * HKVh * Gg * 4 * 4);
  float*    attn  = (float*)carve((size_t)Bsz * HIDn * 4);
  unsigned* aq    = (unsigned*)carve((size_t)Bsz * HIDn);
  float*    as_   = (float*)carve(Bsz * sizeof(float));

  row_quant_kernel<<<Bsz, 256, 0, stream>>>(x, xq, xs, HIDn);
  gemm_w8a8_kernel<<<(Bsz / 16) * (QKVW / 16), 32, 0, stream>>>(
      xq, wqkv, xs, wqs, qkv, Bsz, QKVW, HIDn);
  rope_kv_kernel<<<Bsz * (Hh + 2 * HKVh), 64, 0, stream>>>(qkv, kd, vd, kvs);
  chunk_score_kernel<<<Bsz * HKVh * NCn, 128, 0, stream>>>(kcache, qkv, kvs, score);
  topk_kernel<<<Bsz * HKVh * Gg, 128, 0, stream>>>(score, sel);
  attn_kernel<<<Bsz * HKVh * Gg, 128, 0, stream>>>(
      kcache, vcache, qkv, kd, vd, sel, kvs, attn);
  row_quant_kernel<<<Bsz, 256, 0, stream>>>(attn, aq, as_, HIDn);
  gemm_w8a8_kernel<<<(Bsz / 16) * (HIDn / 16), 32, 0, stream>>>(
      aq, wo, as_, wos, y, Bsz, HIDn, HIDn);
}
